// DenseFullAttention_51642686767693
// MI455X (gfx1250) — matrix operand order repaired and checked by reading, not compiled, not run on hardware
//
#include <hip/hip_runtime.h>

// ---------------------------------------------------------------------------
// Types for CDNA5 WMMA (wave32): V_WMMA_F32_16X16X32_BF16
// ---------------------------------------------------------------------------
typedef __attribute__((ext_vector_type(16))) __bf16 v16bf;
typedef __attribute__((ext_vector_type(8)))  float  v8f;

struct Frag {
    union { v16bf v; unsigned int u[8]; };
};

typedef union { v8f v; float f[8]; } AccF;

union U4 { uint4 q; unsigned short h[8]; };

__device__ __forceinline__ unsigned short f2bf(float f) {
    union { float f; unsigned int u; } x;
    x.f = f;
    unsigned int u = x.u;
    u += 0x7FFFu + ((u >> 16) & 1u);   // round-to-nearest-even
    return (unsigned short)(u >> 16);
}

__device__ __forceinline__ float bf2f(unsigned short h) {
    union { unsigned int u; float f; } x;
    x.u = ((unsigned int)h) << 16;
    return x.f;
}

// ---------------------------------------------------------------------------
// CDNA5 async global->LDS copy (ASYNCcnt path). GVS addressing:
//   mem  = SGPR64 base + VGPR32 byte-offset
//   LDS  = low 32 bits of generic __shared__ pointer (== LDS byte offset,
//          per the ISA's LDS aperture mapping).
// ---------------------------------------------------------------------------
__device__ __forceinline__ void async_load_b128(void* lds_ptr, unsigned goff_bytes,
                                                const void* gbase) {
    unsigned lds_off = (unsigned)(unsigned long long)lds_ptr;
    asm volatile("global_load_async_to_lds_b128 %0, %1, %2"
                 :
                 : "v"(lds_off), "v"(goff_bytes),
                   "s"((unsigned long long)gbase)
                 : "memory");
}

__device__ __forceinline__ void wait_async0() {
    asm volatile("s_wait_asynccnt 0x0" ::: "memory");
}

// A-matrix fragment (16x32, MxK), LDS layout [m][k] row-major, stride even.
__device__ __forceinline__ Frag load_frag_a(const unsigned short* base, int stride,
                                            int row0, int lane) {
    Frag f;
    const unsigned short* p = base + (row0 + (lane & 15)) * stride;
    const int kh = (lane >> 4) << 4;
#pragma unroll
    for (int j = 0; j < 4; ++j) {
        f.u[j]     = *(const unsigned int*)(p + kh + 2 * j);
        f.u[j + 4] = *(const unsigned int*)(p + 8 + kh + 2 * j);
    }
    return f;
}

// B-matrix fragment (32x16, KxN), LDS layout [n][k] (k contiguous), stride even.
__device__ __forceinline__ Frag load_frag_b(const unsigned short* base, int stride,
                                            int col0, int lane) {
    Frag f;
    const unsigned short* p = base + (col0 + (lane & 15)) * stride;
    const int kb = (lane >> 4) << 4;
#pragma unroll
    for (int j = 0; j < 8; ++j)
        f.u[j] = *(const unsigned int*)(p + kb + 2 * j);
    return f;
}

__device__ __forceinline__ v8f wmma_bf16(const Frag& a, const Frag& b, v8f c) {
    return __builtin_amdgcn_wmma_f32_16x16x32_bf16(
        /*neg_a=*/false, a.v, /*neg_b=*/false, b.v,
        /*c_mod=*/(short)0, c, /*reuse_a=*/false, /*reuse_b=*/false);
}

// ---------------------------------------------------------------------------
// Conversion / packing kernels
// ---------------------------------------------------------------------------
__global__ void cvt_f32_bf16_kernel(const float* __restrict__ src,
                                    unsigned short* __restrict__ dst, int n) {
    int i = blockIdx.x * blockDim.x + threadIdx.x;
    int stride = gridDim.x * blockDim.x;
    for (; i < n; i += stride) dst[i] = f2bf(src[i]);
}

__global__ void pack_wqkv_kernel(const float* __restrict__ wq,
                                 const float* __restrict__ wk,
                                 const float* __restrict__ wv,
                                 unsigned short* __restrict__ dst) {
    const int total = 2048 * 6144;
    int i = blockIdx.x * blockDim.x + threadIdx.x;
    int stride = gridDim.x * blockDim.x;
    for (; i < total; i += stride) {
        int k = i / 6144, n = i - k * 6144;
        int sel = n >> 11, col = n & 2047;
        const float* s = (sel == 0) ? wq : (sel == 1 ? wk : wv);
        dst[i] = f2bf(s[k * 2048 + col]);
    }
}

// ---------------------------------------------------------------------------
// BF16 WMMA GEMM: C[M,N] = A[M,K] * B[K,N], block tile 128x128, BK=32,
// 256 threads = 8 wave32 waves, wave tile 64x32 = 4x2 WMMA tiles.
// A tile staged via async global->LDS; B tile transposed manually.
// ---------------------------------------------------------------------------
#define GLDA 40
#define GLDB 40

template <int F32OUT>
__global__ __launch_bounds__(256) void gemm_bf16_kernel(
    const unsigned short* __restrict__ A, const unsigned short* __restrict__ B,
    void* __restrict__ Cv, int M, int N, int K, int lda, int ldb, int ldc) {
    __shared__ __align__(16) unsigned short sA[128 * GLDA];   // [m][k]
    __shared__ __align__(16) unsigned short sB[128 * GLDB];   // [n][k]

    const int tid  = threadIdx.x;
    const int lane = tid & 31;
    const int w    = tid >> 5;
    const int wm   = w >> 2;        // 0..1
    const int wn   = w & 3;         // 0..3
    const int row0 = blockIdx.y * 128;
    const int col0 = blockIdx.x * 128;

    AccF c[4][2];
#pragma unroll
    for (int i = 0; i < 4; ++i)
#pragma unroll
        for (int j = 0; j < 2; ++j)
#pragma unroll
            for (int v = 0; v < 8; ++v) c[i][j].f[v] = 0.f;

    for (int k0 = 0; k0 < K; k0 += 32) {
        __syncthreads();
        // A tile: 128x32, async global->LDS (no VGPR round trip)
        const unsigned short* Abase = A + (size_t)row0 * lda + k0;
#pragma unroll
        for (int i = 0; i < 2; ++i) {
            int idx = tid + i * 256;
            int r = idx >> 2, cc = (idx & 3) * 8;
            async_load_b128(sA + r * GLDA + cc,
                            (unsigned)((r * lda + cc) * 2), Abase);
        }
        // B tile: 32x128, coalesced uint4 loads, scatter-transposed to [n][k]
#pragma unroll
        for (int i = 0; i < 2; ++i) {
            int idx = tid + i * 256;
            int kr = idx >> 4, nc = (idx & 15) * 8;
            U4 u;
            u.q = *(const uint4*)(B + (size_t)(k0 + kr) * ldb + col0 + nc);
#pragma unroll
            for (int e = 0; e < 8; ++e) sB[(nc + e) * GLDB + kr] = u.h[e];
        }
        if (k0 + 32 < K) {
            __builtin_prefetch(A + (size_t)(row0 + (tid & 127)) * lda + k0 + 32, 0, 0);
            __builtin_prefetch(B + (size_t)(k0 + 32 + (tid & 31)) * ldb + col0, 0, 0);
        }
        wait_async0();
        __syncthreads();

        Frag bfr[2];
#pragma unroll
        for (int j = 0; j < 2; ++j)
            bfr[j] = load_frag_b(sB, GLDB, wn * 32 + j * 16, lane);
#pragma unroll
        for (int i = 0; i < 4; ++i) {
            Frag a = load_frag_a(sA, GLDA, wm * 64 + i * 16, lane);
#pragma unroll
            for (int j = 0; j < 2; ++j)
                c[i][j].v = wmma_bf16(a, bfr[j], c[i][j].v);
        }
    }

    const int half = lane >> 4, nloc = lane & 15;
#pragma unroll
    for (int i = 0; i < 4; ++i)
#pragma unroll
        for (int j = 0; j < 2; ++j)
#pragma unroll
            for (int v = 0; v < 8; ++v) {
                int m = row0 + wm * 64 + i * 16 + v + half * 8;
                int n = col0 + wn * 32 + j * 16 + nloc;
                if (F32OUT)
                    ((float*)Cv)[(size_t)m * ldc + n] = c[i][j].f[v];
                else
                    ((unsigned short*)Cv)[(size_t)m * ldc + n] = f2bf(c[i][j].f[v]);
            }
}

// ---------------------------------------------------------------------------
// RoPE on Q and K halves of the packed QKV buffer ([8192][6144] bf16)
// ---------------------------------------------------------------------------
__global__ void rope_kernel(unsigned short* __restrict__ qkv,
                            const float* __restrict__ cs,
                            const float* __restrict__ sn) {
    int idx = blockIdx.x * blockDim.x + threadIdx.x;  // B*S*H*64 = 8388608
    int d = idx & 63;
    int h = (idx >> 6) & 15;
    int s = (idx >> 10) & 4095;
    int b = idx >> 22;
    float co = cs[s * 64 + d], si = sn[s * 64 + d];
    size_t m = (size_t)b * 4096 + s;
#pragma unroll
    for (int part = 0; part < 2; ++part) {
        size_t base = m * 6144 + (size_t)part * 2048 + h * 128;
        float x1 = bf2f(qkv[base + d]);
        float x2 = bf2f(qkv[base + d + 64]);
        qkv[base + d]      = f2bf(x1 * co - x2 * si);
        qkv[base + d + 64] = f2bf(x2 * co + x1 * si);
    }
}

// ---------------------------------------------------------------------------
// Flash-style causal attention. Grid: (S/128, B*H). 256 threads = 8 waves.
// Q tile 128x128 resident in LDS (async-loaded); stream 32-key tiles of K/V
// (K async, V manual transpose). S = Q*K^T and O += P*V via bf16 WMMA.
// ---------------------------------------------------------------------------
#define ALDQ 136
#define ALDK 136
#define ALDV 40
#define ALDP 40

__global__ __launch_bounds__(256) void attn_kernel(
    const unsigned short* __restrict__ qkv, unsigned short* __restrict__ out) {
    __shared__ __align__(16) unsigned short sQ[128 * ALDQ];   // [q][d]
    __shared__ __align__(16) unsigned short sK[32 * ALDK];    // [key][d] == B-layout [n][k]
    __shared__ __align__(16) unsigned short sV[128 * ALDV];   // [d][key] == B-layout [n][k]
    __shared__ __align__(16) unsigned short sP[8 * 16 * ALDP];// per-wave [q][key] A-layout

    const int tid  = threadIdx.x;
    const int lane = tid & 31;
    const int w    = tid >> 5;
    const int half = lane >> 4;
    const int nloc = lane & 15;
    const int b    = blockIdx.y >> 4;
    const int h    = blockIdx.y & 15;
    const int q0   = blockIdx.x * 128;
    const long m0  = (long)b * 4096 + q0;
    const long ld  = 6144;

    const unsigned short* Qg = qkv + m0 * ld + h * 128;
    const unsigned short* Kg = qkv + (long)b * 4096 * ld + 2048 + h * 128;
    const unsigned short* Vg = qkv + (long)b * 4096 * ld + 4096 + h * 128;

    // Load the 128x128 Q tile once (async copy; completion covered by the
    // s_wait_asynccnt before the first compute barrier).
#pragma unroll
    for (int i = 0; i < 8; ++i) {
        int idx = tid + i * 256;
        int r = idx >> 4, cc = (idx & 15) * 8;
        async_load_b128(sQ + r * ALDQ + cc, (unsigned)((r * ld + cc) * 2), Qg);
    }

    AccF oc[8];
    float mrow[8], lrow[8];
#pragma unroll
    for (int ot = 0; ot < 8; ++ot)
#pragma unroll
        for (int v = 0; v < 8; ++v) oc[ot].f[v] = 0.f;
#pragma unroll
    for (int v = 0; v < 8; ++v) { mrow[v] = -1e30f; lrow[v] = 0.f; }

    const float scale = 0.08838834764831845f;  // 1/sqrt(128)
    const int ktmax = (q0 + 127) >> 5;

    for (int kt = 0; kt <= ktmax; ++kt) {
        const int k0 = kt * 32;
        __syncthreads();
        // K tile 32x128 -> sK row-major (async copy)
        const unsigned short* Kbase = Kg + (long)k0 * ld;
#pragma unroll
        for (int i = 0; i < 2; ++i) {
            int idx = tid + i * 256;
            int r = idx >> 4, cc = (idx & 15) * 8;
            async_load_b128(sK + r * ALDK + cc, (unsigned)((r * ld + cc) * 2), Kbase);
        }
        // V tile 32x128 -> sV transposed [d][key] (manual)
#pragma unroll
        for (int i = 0; i < 2; ++i) {
            int idx = tid + i * 256;
            int r = idx >> 4, cc = (idx & 15) * 8;
            U4 u;
            u.q = *(const uint4*)(Vg + (long)(k0 + r) * ld + cc);
#pragma unroll
            for (int e = 0; e < 8; ++e) sV[(cc + e) * ALDV + r] = u.h[e];
        }
        wait_async0();
        __syncthreads();

        // S = Q * K^T : wave rows w*16..w*16+15, 2 col-tiles of 16 keys
        AccF sacc[2];
#pragma unroll
        for (int t = 0; t < 2; ++t)
#pragma unroll
            for (int v = 0; v < 8; ++v) sacc[t].f[v] = 0.f;
#pragma unroll
        for (int kk = 0; kk < 128; kk += 32) {
            Frag a = load_frag_a(sQ + kk, ALDQ, w * 16, lane);
#pragma unroll
            for (int t = 0; t < 2; ++t) {
                Frag bk = load_frag_b(sK + kk, ALDK, t * 16, lane);
                sacc[t].v = wmma_bf16(a, bk, sacc[t].v);
            }
        }

        // Scale, causal mask, online softmax (row stats via shfl_xor in a half)
        float alpha[8];
#pragma unroll
        for (int v = 0; v < 8; ++v) {
            const int qpos = q0 + w * 16 + v + half * 8;
#pragma unroll
            for (int t = 0; t < 2; ++t) {
                float x = sacc[t].f[v] * scale;
                int kpos = k0 + t * 16 + nloc;
                if (kpos > qpos) x = -1e30f;
                sacc[t].f[v] = x;
            }
            float rm = fmaxf(sacc[0].f[v], sacc[1].f[v]);
            rm = fmaxf(rm, __shfl_xor(rm, 1));
            rm = fmaxf(rm, __shfl_xor(rm, 2));
            rm = fmaxf(rm, __shfl_xor(rm, 4));
            rm = fmaxf(rm, __shfl_xor(rm, 8));
            float mnew = fmaxf(mrow[v], rm);
            alpha[v] = __expf(mrow[v] - mnew);
            mrow[v] = mnew;
            float p0 = __expf(sacc[0].f[v] - mnew);
            float p1 = __expf(sacc[1].f[v] - mnew);
            sacc[0].f[v] = p0;
            sacc[1].f[v] = p1;
            float rs = p0 + p1;
            rs += __shfl_xor(rs, 1);
            rs += __shfl_xor(rs, 2);
            rs += __shfl_xor(rs, 4);
            rs += __shfl_xor(rs, 8);
            lrow[v] = lrow[v] * alpha[v] + rs;
        }
        // Rescale running O
#pragma unroll
        for (int ot = 0; ot < 8; ++ot)
#pragma unroll
            for (int v = 0; v < 8; ++v) oc[ot].f[v] *= alpha[v];

        // Stage P (16x32 bf16) through per-wave LDS to get A-fragment layout
        unsigned short* myP = sP + w * 16 * ALDP;
#pragma unroll
        for (int t = 0; t < 2; ++t)
#pragma unroll
            for (int v = 0; v < 8; ++v)
                myP[(v + half * 8) * ALDP + t * 16 + nloc] = f2bf(sacc[t].f[v]);

        // O += P * V (8 dim-tiles of 16)
        Frag pa = load_frag_a(myP, ALDP, 0, lane);
#pragma unroll
        for (int ot = 0; ot < 8; ++ot) {
            Frag vb = load_frag_b(sV, ALDV, ot * 16, lane);
            oc[ot].v = wmma_bf16(pa, vb, oc[ot].v);
        }
    }

    // Normalize and store bf16 to attn buffer [8192][2048]
    float inv[8];
#pragma unroll
    for (int v = 0; v < 8; ++v) inv[v] = 1.0f / lrow[v];
    unsigned short* op = out + (size_t)(m0 + w * 16) * 2048 + h * 128;
#pragma unroll
    for (int ot = 0; ot < 8; ++ot)
#pragma unroll
        for (int v = 0; v < 8; ++v)
            op[(size_t)(v + half * 8) * 2048 + ot * 16 + nloc] =
                f2bf(oc[ot].f[v] * inv[v]);
}

// ---------------------------------------------------------------------------
// Host launcher
// ---------------------------------------------------------------------------
extern "C" void kernel_launch(void* const* d_in, const int* in_sizes, int n_in,
                              void* d_out, int out_size, void* d_ws, size_t ws_size,
                              hipStream_t stream) {
    (void)in_sizes; (void)n_in; (void)out_size; (void)ws_size;
    const float* x    = (const float*)d_in[0];
    const float* wq   = (const float*)d_in[1];
    const float* wk   = (const float*)d_in[2];
    const float* wv   = (const float*)d_in[3];
    const float* wo   = (const float*)d_in[4];
    const float* cosb = (const float*)d_in[5];
    const float* sinb = (const float*)d_in[6];

    char* ws = (char*)d_ws;
    unsigned short* xb   = (unsigned short*)(ws);                 // 8192x2048 bf16
    unsigned short* wqkv = (unsigned short*)(ws + 33554432);      // 2048x6144 bf16
    unsigned short* wob  = (unsigned short*)(ws + 58720256);      // 2048x2048 bf16
    unsigned short* qkv  = (unsigned short*)(ws + 67108864);      // 8192x6144 bf16
    unsigned short* attn = (unsigned short*)(ws + 167772160);     // 8192x2048 bf16

    cvt_f32_bf16_kernel<<<8192, 256, 0, stream>>>(x, xb, 8192 * 2048);
    pack_wqkv_kernel<<<8192, 256, 0, stream>>>(wq, wk, wv, wqkv);
    cvt_f32_bf16_kernel<<<4096, 256, 0, stream>>>(wo, wob, 2048 * 2048);

    // Fused QKV projection: [8192,2048] x [2048,6144] -> qkv (bf16)
    gemm_bf16_kernel<0><<<dim3(48, 64), 256, 0, stream>>>(
        xb, wqkv, (void*)qkv, 8192, 6144, 2048, 2048, 6144, 6144);

    rope_kernel<<<32768, 256, 0, stream>>>(qkv, cosb, sinb);

    attn_kernel<<<dim3(32, 32), 256, 0, stream>>>(qkv, attn);

    // Output projection: [8192,2048] x [2048,2048] -> d_out (fp32)
    gemm_bf16_kernel<1><<<dim3(16, 64), 256, 0, stream>>>(
        attn, wob, d_out, 8192, 2048, 2048, 2048, 2048, 2048);
}
